// VectorQuantizer_77867757076524
// MI455X (gfx1250) — matrix operand order repaired and checked by reading, not compiled, not run on hardware
//
#include <hip/hip_runtime.h>
#include <hip/hip_bf16.h>

typedef __attribute__((ext_vector_type(16))) _Float16 v16h;
typedef __attribute__((ext_vector_type(8)))  _Float16 v8h;
typedef __attribute__((ext_vector_type(8)))  float    v8f;

#define DIM 32            // feature dim
#define KCODES 256        // codebook entries
#define NTILES 16         // KCODES / 16
#define WAVES 8           // waves per block
#define TOK_PER_WAVE 16
#define TOK_PER_BLOCK 128 // WAVES * TOK_PER_WAVE

__launch_bounds__(256, 1)
__global__ void vq_wmma_kernel(const float* __restrict__ x,
                               const float* __restrict__ cb,
                               int*   __restrict__ idx_out,
                               float* __restrict__ probs,
                               float* __restrict__ quants,
                               float* __restrict__ loss,
                               int Ntok)
{
    // LDS: 32K fp32 codebook + 16K f16 codebook + 1K norms + 8K x tiles + 0.5K x norms ~= 57.5 KB
    __shared__ __align__(64) float    cbf[KCODES * DIM];
    __shared__ __align__(64) _Float16 cbh[KCODES * DIM];
    __shared__ __align__(64) float    cnorm_s[KCODES];
    __shared__ __align__(64) _Float16 xa[WAVES * TOK_PER_WAVE * DIM];
    __shared__ __align__(64) float    xnorm_s[WAVES * TOK_PER_WAVE];

    const int tid  = threadIdx.x;
    const int wave = tid >> 5;
    const int lane = tid & 31;
    const int g    = lane >> 4;   // half-wave group (selects K-slice / token rows 8g..8g+7)
    const int n    = lane & 15;   // code column within a 16x16 tile

    // ---- stage codebook: thread t -> code row t (blockDim == KCODES) ----
    {
        const float* crow = cb + tid * DIM;
        float cn = 0.f;
        #pragma unroll
        for (int j = 0; j < DIM; j += 4) {
            float4 v = *(const float4*)(crow + j);
            cn += v.x * v.x + v.y * v.y + v.z * v.z + v.w * v.w;
            cbf[tid * DIM + j + 0] = v.x;
            cbf[tid * DIM + j + 1] = v.y;
            cbf[tid * DIM + j + 2] = v.z;
            cbf[tid * DIM + j + 3] = v.w;
            cbh[tid * DIM + j + 0] = (_Float16)v.x;
            cbh[tid * DIM + j + 1] = (_Float16)v.y;
            cbh[tid * DIM + j + 2] = (_Float16)v.z;
            cbh[tid * DIM + j + 3] = (_Float16)v.w;
        }
        cnorm_s[tid] = cn;
    }

    // ---- stage this wave's 16 tokens: lane handles token row n, dim-half g ----
    const int tb = blockIdx.x * TOK_PER_BLOCK + wave * TOK_PER_WAVE;
    {
        const float* xrow = x + (size_t)(tb + n) * DIM + 16 * g;
        _Float16* xdst = xa + (wave * TOK_PER_WAVE + n) * DIM + 16 * g;
        float pn = 0.f;
        #pragma unroll
        for (int j = 0; j < 16; j += 4) {
            float4 v = *(const float4*)(xrow + j);
            pn += v.x * v.x + v.y * v.y + v.z * v.z + v.w * v.w;
            xdst[j + 0] = (_Float16)v.x;
            xdst[j + 1] = (_Float16)v.y;
            xdst[j + 2] = (_Float16)v.z;
            xdst[j + 3] = (_Float16)v.w;
        }
        float full = pn + __shfl_xor(pn, 16, 32);
        if (g == 0) xnorm_s[wave * TOK_PER_WAVE + n] = full;
    }
    __syncthreads();

    // ---- A fragment (16x32 f16): lane (g,m=n): K = 16g + e ----
    v16h afrag;
    {
        const _Float16* xw = xa + (wave * TOK_PER_WAVE + n) * DIM;
        v8h alo = *(const v8h*)(xw + 16 * g);
        v8h ahi = *(const v8h*)(xw + 16 * g + 8);
        #pragma unroll
        for (int i = 0; i < 8; i++) { afrag[i] = alo[i]; afrag[i + 8] = ahi[i]; }
    }

    // per-lane token-row norms (rows M = 8g + r)
    float xn[8];
    #pragma unroll
    for (int r = 0; r < 8; r++) xn[r] = xnorm_s[wave * TOK_PER_WAVE + 8 * g + r];

    // ---- pass A: online softmax stats + argmax over this lane's 16 codes per row ----
    float rmax[8], rsum[8], rbv[8];
    int   rbi[8];
    #pragma unroll
    for (int r = 0; r < 8; r++) { rmax[r] = -3.0e38f; rsum[r] = 0.f; rbv[r] = -3.0e38f; rbi[r] = 0; }

    #pragma unroll
    for (int t = 0; t < NTILES; t++) {
        const int c = t * 16 + n;
        v16h bfrag = *(const v16h*)(cbh + c * DIM + 16 * g);  // B 32x16: K = e + 16g
        float cn = cnorm_s[c];
        v8f acc = {};
        acc = __builtin_amdgcn_wmma_f32_16x16x32_f16(false, afrag, false, bfrag,
                                                     (short)0, acc, false, false);
        #pragma unroll
        for (int r = 0; r < 8; r++) {
            float logit = 2.f * acc[r] - xn[r] - cn;   // = -d2
            if (logit > rbv[r]) { rbv[r] = logit; rbi[r] = c; }
            float nm = fmaxf(rmax[r], logit);
            rsum[r] = rsum[r] * __expf(rmax[r] - nm) + __expf(logit - nm);
            rmax[r] = nm;
        }
    }

    // ---- reduce (max,sum) and (argmax) across the 16 lanes of the half-group ----
    #pragma unroll
    for (int mask = 1; mask < 16; mask <<= 1) {
        #pragma unroll
        for (int r = 0; r < 8; r++) {
            float om = __shfl_xor(rmax[r], mask, 32);
            float os = __shfl_xor(rsum[r], mask, 32);
            float nm = fmaxf(rmax[r], om);
            rsum[r] = rsum[r] * __expf(rmax[r] - nm) + os * __expf(om - nm);
            rmax[r] = nm;
            float obv = __shfl_xor(rbv[r], mask, 32);
            int   obi = __shfl_xor(rbi[r], mask, 32);
            if (obv > rbv[r] || (obv == rbv[r] && obi < rbi[r])) { rbv[r] = obv; rbi[r] = obi; }
        }
    }
    float rinv[8];
    #pragma unroll
    for (int r = 0; r < 8; r++) rinv[r] = 1.f / rsum[r];

    // ---- pass B: recompute tiles (compute is ~free), write normalized probs ----
    #pragma unroll
    for (int t = 0; t < NTILES; t++) {
        const int c = t * 16 + n;
        v16h bfrag = *(const v16h*)(cbh + c * DIM + 16 * g);
        float cn = cnorm_s[c];
        v8f acc = {};
        acc = __builtin_amdgcn_wmma_f32_16x16x32_f16(false, afrag, false, bfrag,
                                                     (short)0, acc, false, false);
        #pragma unroll
        for (int r = 0; r < 8; r++) {
            float logit = 2.f * acc[r] - xn[r] - cn;
            int tok = tb + 8 * g + r;
            probs[(size_t)tok * KCODES + c] = __expf(logit - rmax[r]) * rinv[r];
        }
    }

    // ---- indices (one lane per half-group writes its 8 token rows) ----
    if (n == 0) {
        #pragma unroll
        for (int r = 0; r < 8; r++) idx_out[tb + 8 * g + r] = rbi[r];
    }

    // ---- quants (straight-through == gathered codebook row) + loss ----
    {
        const int row  = n >> 1;           // token row 0..7 within half-group
        const int half = n & 1;            // dim half
        const int tok  = tb + 8 * g + row;
        const int code = rbi[row];
        const float* qrow = cbf + code * DIM + 16 * half;
        const float* xrow = x + (size_t)tok * DIM + 16 * half;
        float*       qout = quants + (size_t)tok * DIM + 16 * half;
        float pl = 0.f;
        #pragma unroll
        for (int j = 0; j < 16; j += 4) {
            float4 q  = *(const float4*)(qrow + j);
            float4 xv = *(const float4*)(xrow + j);
            *(float4*)(qout + j) = q;
            float dx = q.x - xv.x, dy = q.y - xv.y, dz = q.z - xv.z, dw = q.w - xv.w;
            pl += dx * dx + dy * dy + dz * dz + dw * dw;
        }
        float tot = pl + __shfl_xor(pl, 1, 32);
        if (half == 0) loss[tok] = 1.25f * tot * (1.0f / 32.0f);  // (1+MU)*mean, MU=0.25
    }
}

extern "C" void kernel_launch(void* const* d_in, const int* in_sizes, int n_in,
                              void* d_out, int out_size, void* d_ws, size_t ws_size,
                              hipStream_t stream) {
    const float* x  = (const float*)d_in[0];   // [32,4096,32] f32
    const float* cb = (const float*)d_in[1];   // [256,32] f32
    const int Ntok = in_sizes[0] / DIM;        // 131072

    // d_out = concat(indices[N] (i32 bits), probs[N*256], quants[N*32], loss[N])
    float* out    = (float*)d_out;
    int*   idx_o  = (int*)out;
    float* probs  = out + Ntok;
    float* quants = probs + (size_t)Ntok * KCODES;
    float* lossp  = quants + (size_t)Ntok * DIM;

    const int blocks = Ntok / TOK_PER_BLOCK;   // 1024
    vq_wmma_kernel<<<blocks, 256, 0, stream>>>(x, cb, idx_o, probs, quants, lossp, Ntok);
}
